// SwinTransformerBlock_3D_11287174054495
// MI455X (gfx1250) — compile-verified, matrix operand order checked by reading, not run on hardware
//
#include <hip/hip_runtime.h>
#include <hip/hip_bf16.h>
#include <math.h>

// ---------------- problem constants ----------------
#define VV   4
#define BB   2
#define HH   32
#define WW   64
#define ZZ   13
#define CC   256
#define NHD  8
#define HD   32            // CC / NHD
#define HID_ 1024
#define LL   (HH*WW*ZZ)    // 26624
#define NWH  (HH/4)        // 8
#define NWW  (WW/4)        // 16
#define BWIN (BB*NWH*NWW*ZZ) // 3328 windows per variable
#define MROWS (BWIN*16)      // 53248 window-token rows per variable
#define TOKENS ((long)VV*BB*LL) // 212992

typedef _Float16 v16h __attribute__((ext_vector_type(16)));
typedef _Float16 h8   __attribute__((ext_vector_type(8)));
typedef float    v8f  __attribute__((ext_vector_type(8)));

// Build a 16x32 f16 A/B WMMA fragment from a row-major K-contiguous row.
// Lane mapping (per CDNA5 ISA 7.12.2): elements 0..7 -> K = kbase..kbase+7,
// elements 8..15 -> K = kbase+16..kbase+23, kbase = (lane>>4)*8.
__device__ __forceinline__ v16h load_frag16(const _Float16* __restrict__ p, int kbase) {
  h8 lo = *(const h8*)(p + kbase);
  h8 hi = *(const h8*)(p + kbase + 16);
  v16h r;
#pragma unroll
  for (int i = 0; i < 8; ++i) { r[i] = lo[i]; r[i + 8] = hi[i]; }
  return r;
}

#define WMMA_F16(A_, B_, C_) \
  __builtin_amdgcn_wmma_f32_16x16x32_f16(false, (A_), false, (B_), (short)0, (C_), false, false)

// Register-blocked GEMM core: one wave computes a 32(M) x 64(N) output tile.
// 8 WMMAs per K-step from 2 A-fragments + 4 B-fragments (1.5 b128 loads / WMMA).
// acc[t*4+j] = tile (M sub-tile t in {0,1}, N sub-tile j in {0..3}).
__device__ __forceinline__ void gemm_core_32x64(
    const _Float16* __restrict__ a0p, const _Float16* __restrict__ a1p,
    const _Float16* __restrict__ b0p, const _Float16* __restrict__ b1p,
    const _Float16* __restrict__ b2p, const _Float16* __restrict__ b3p,
    int K, int kbase, v8f acc[8]) {
  for (int k0 = 0; k0 < K; k0 += 32) {
    v16h a0 = load_frag16(a0p + k0, kbase);
    v16h a1 = load_frag16(a1p + k0, kbase);
    v16h b0 = load_frag16(b0p + k0, kbase);
    v16h b1 = load_frag16(b1p + k0, kbase);
    v16h b2 = load_frag16(b2p + k0, kbase);
    v16h b3 = load_frag16(b3p + k0, kbase);
    acc[0] = WMMA_F16(a0, b0, acc[0]);
    acc[1] = WMMA_F16(a0, b1, acc[1]);
    acc[2] = WMMA_F16(a0, b2, acc[2]);
    acc[3] = WMMA_F16(a0, b3, acc[3]);
    acc[4] = WMMA_F16(a1, b0, acc[4]);
    acc[5] = WMMA_F16(a1, b1, acc[5]);
    acc[6] = WMMA_F16(a1, b2, acc[6]);
    acc[7] = WMMA_F16(a1, b3, acc[7]);
  }
}

// ---------------- kernel 0: weight transpose + f32->f16 ----------------
// in:  w  [V,K,N] f32 ; out: wt [V,N,K] f16
__global__ __launch_bounds__(256) void k_wt(const float* __restrict__ w,
                                            _Float16* __restrict__ wt,
                                            int K, int N) {
  long i = (long)blockIdx.x * 256 + threadIdx.x;
  long tot = (long)VV * K * N;
  if (i >= tot) return;
  int v = (int)(i / ((long)K * N));
  int r = (int)(i % ((long)K * N));
  int k = r / N, n = r % N;
  wt[((size_t)v * N + n) * K + k] = (_Float16)w[i];
}

// ---------------- kernel 1: LN1 + cyclic shift + window partition ----------------
// One wave per token; lane handles 8 channels. Output f16 [V][MROWS][CC].
__global__ __launch_bounds__(256) void k_ln1_window(
    const float* __restrict__ x0, const float* __restrict__ x1,
    const float* __restrict__ x2, const float* __restrict__ x3,
    const float* __restrict__ g, const float* __restrict__ b,
    _Float16* __restrict__ out) {
  int wid = threadIdx.x >> 5, lane = threadIdx.x & 31;
  long t = (long)blockIdx.x * 8 + wid;          // 0..TOKENS-1
  int v   = (int)(t / (BB * (long)LL));
  int rem = (int)(t % (BB * (long)LL));
  int bb  = rem / LL;
  int l   = rem % LL;
  const float* xp = (v == 0) ? x0 : (v == 1) ? x1 : (v == 2) ? x2 : x3;
  const float* row = xp + ((size_t)bb * LL + l) * CC;
  float vals[8]; float s = 0.f;
#pragma unroll
  for (int i = 0; i < 8; ++i) { vals[i] = row[lane * 8 + i]; s += vals[i]; }
#pragma unroll
  for (int m = 16; m >= 1; m >>= 1) s += __shfl_xor(s, m, 32);
  float mu = s * (1.f / CC);
  float var = 0.f;
#pragma unroll
  for (int i = 0; i < 8; ++i) { float d = vals[i] - mu; var += d * d; }
#pragma unroll
  for (int m = 16; m >= 1; m >>= 1) var += __shfl_xor(var, m, 32);
  float rs = rsqrtf(var * (1.f / CC) + 1e-5f);
  // shifted + windowed destination
  int h = l / (WW * ZZ), w = (l / ZZ) % WW, z = l % ZZ;
  int h2 = (h + 2) & (HH - 1), w2 = (w + 2) & (WW - 1);
  int b_ = ((bb * NWH + (h2 >> 2)) * NWW + (w2 >> 2)) * ZZ + z;
  int n  = ((h2 & 3) << 2) | (w2 & 3);
  _Float16* orow = out + ((size_t)v * MROWS + (size_t)b_ * 16 + n) * CC;
  const float* gg = g + v * CC;
  const float* bbp = b + v * CC;
#pragma unroll
  for (int i = 0; i < 8; ++i) {
    int c = lane * 8 + i;
    orow[c] = (_Float16)((vals[i] - mu) * rs * gg[c] + bbp[c]);
  }
}

// ---------------- kernel: LN2 (natural token layout) ----------------
__global__ __launch_bounds__(256) void k_ln2(const float* __restrict__ x2,
                                             const float* __restrict__ g,
                                             const float* __restrict__ b,
                                             _Float16* __restrict__ out) {
  int wid = threadIdx.x >> 5, lane = threadIdx.x & 31;
  long t = (long)blockIdx.x * 8 + wid;
  int v = (int)(t / (BB * (long)LL));
  const float* row = x2 + (size_t)t * CC;
  float vals[8]; float s = 0.f;
#pragma unroll
  for (int i = 0; i < 8; ++i) { vals[i] = row[lane * 8 + i]; s += vals[i]; }
#pragma unroll
  for (int m = 16; m >= 1; m >>= 1) s += __shfl_xor(s, m, 32);
  float mu = s * (1.f / CC);
  float var = 0.f;
#pragma unroll
  for (int i = 0; i < 8; ++i) { float d = vals[i] - mu; var += d * d; }
#pragma unroll
  for (int m = 16; m >= 1; m >>= 1) var += __shfl_xor(var, m, 32);
  float rs = rsqrtf(var * (1.f / CC) + 1e-5f);
  const float* gg = g + v * CC;
  const float* bbp = b + v * CC;
  _Float16* orow = out + (size_t)t * CC;
#pragma unroll
  for (int i = 0; i < 8; ++i) {
    int c = lane * 8 + i;
    orow[c] = (_Float16)((vals[i] - mu) * rs * gg[c] + bbp[c]);
  }
}

// ---------------- generic WMMA GEMM, f16 out, optional exact GELU ----------------
// A [M,K] f16, Bt [N,K] f16, bias [N] f32 ; per-variable strides via blockIdx.z.
// Workgroup tile 64(M) x 256(N): 8 waves as 2(M) x 4(N), each wave 32x64.
__global__ __launch_bounds__(256) void k_gemm_f16(
    const _Float16* __restrict__ A, const _Float16* __restrict__ Bt,
    const float* __restrict__ bias, _Float16* __restrict__ Out,
    int M, int N, int K, int act,
    size_t sAv, size_t sBv, size_t sBiasv, size_t sOutv) {
  int v = blockIdx.z;
  A += (size_t)v * sAv; Bt += (size_t)v * sBv;
  bias += (size_t)v * sBiasv; Out += (size_t)v * sOutv;
  int wid = threadIdx.x >> 5, lane = threadIdx.x & 31;
  int mw = wid >> 2, nw = wid & 3;
  int m0 = blockIdx.y * 64 + mw * 32;
  int n0 = blockIdx.x * 256 + nw * 64;
  int lm = lane & 15, hi = lane >> 4, kbase = hi * 8;
  const _Float16* a0p = A + (size_t)(m0 + lm) * K;
  const _Float16* a1p = a0p + (size_t)16 * K;
  const _Float16* b0p = Bt + (size_t)(n0 + lm) * K;
  v8f acc[8];
#pragma unroll
  for (int i = 0; i < 8; ++i) acc[i] = (v8f){0.f,0.f,0.f,0.f,0.f,0.f,0.f,0.f};
  gemm_core_32x64(a0p, a1p, b0p, b0p + 16 * K, b0p + 32 * K, b0p + 48 * K, K, kbase, acc);
#pragma unroll
  for (int t = 0; t < 2; ++t) {
#pragma unroll
    for (int j = 0; j < 4; ++j) {
      int col = n0 + j * 16 + lm;
      float bs = bias[col];
      v8f a = acc[t * 4 + j];
#pragma unroll
      for (int i = 0; i < 8; ++i) {
        int row = m0 + t * 16 + hi * 8 + i;
        float x = a[i] + bs;
        if (act) x = 0.5f * x * (1.f + erff(x * 0.70710678118654752f));
        Out[(size_t)row * N + col] = (_Float16)x;
      }
    }
  }
}

// ---------------- attention: one wave per (window, variable, head) ----------------
// qkv f16 [V][MROWS][3*CC]; ctx f16 [V][MROWS][CC] with the reference's (n,v) row interleave.
__global__ __launch_bounds__(256) void k_attn(const _Float16* __restrict__ qkv,
                                              _Float16* __restrict__ ctx) {
  __shared__ float sp[8][16][16];
  int wid = threadIdx.x >> 5, lane = threadIdx.x & 31;
  long gid = (long)blockIdx.x * 8 + wid;        // 0..BWIN*VV*NHD-1
  int head = (int)(gid & 7);
  int vi   = (int)((gid >> 3) & 3);
  int b_   = (int)(gid >> 5);
  const _Float16* base = qkv + ((size_t)vi * MROWS + (size_t)b_ * 16) * (3 * CC);
  int lm = lane & 15, hif = lane >> 4, kbase = hif * 8;
  const float scale = 0.17677669529663687f;      // hd^-0.5, hd=32

  // q fragment (A, 16x32): row = token lm, K = head-dim, scale folded in
  v16h qa;
  {
    const _Float16* qrow = base + (size_t)lm * (3 * CC) + head * HD;
    h8 lo = *(const h8*)(qrow + kbase);
    h8 hi2 = *(const h8*)(qrow + kbase + 16);
#pragma unroll
    for (int i = 0; i < 8; ++i) {
      qa[i]     = (_Float16)((float)lo[i] * scale);
      qa[i + 8] = (_Float16)((float)hi2[i] * scale);
    }
  }
  // k fragment (B, 32x16): output column = key token lm
  const _Float16* krow = base + (size_t)lm * (3 * CC) + CC + head * HD;
  v16h kb = load_frag16(krow, kbase);
  v8f zero = {0.f,0.f,0.f,0.f,0.f,0.f,0.f,0.f};
  v8f s = WMMA_F16(qa, kb, zero);

  // softmax across the 16 keys (scores for one query row live across a 16-lane half)
  float p[8];
#pragma unroll
  for (int i = 0; i < 8; ++i) {
    float sc = s[i];
    float mx = sc;
#pragma unroll
    for (int m = 8; m >= 1; m >>= 1) mx = fmaxf(mx, __shfl_xor(mx, m, 32));
    float e = __expf(sc - mx);
    float sm = e;
#pragma unroll
    for (int m = 8; m >= 1; m >>= 1) sm += __shfl_xor(sm, m, 32);
    p[i] = e / sm;
  }
  // transpose probs through LDS into A-fragment orientation
#pragma unroll
  for (int i = 0; i < 8; ++i) sp[wid][hif * 8 + i][lm] = p[i];
  __syncthreads();
  v16h pa;
#pragma unroll
  for (int e = 0; e < 8; ++e) {
    pa[e]     = (_Float16)sp[wid][lm][kbase + e]; // K in [0,16)
    pa[e + 8] = (_Float16)0.0f;                   // K in [16,32) zero-pad
  }
  // ctx = P(16x32, K padded) x V(32x16) per 16-wide hd segment
#pragma unroll
  for (int seg = 0; seg < 2; ++seg) {
    int d = seg * 16 + lm;
    const _Float16* vcol = base + 2 * CC + head * HD + d;
    v16h vb;
#pragma unroll
    for (int e = 0; e < 8; ++e) {
      vb[e]     = vcol[(size_t)(kbase + e) * (3 * CC)];
      vb[e + 8] = (_Float16)0.0f;
    }
    v8f c0 = WMMA_F16(pa, vb, zero);
#pragma unroll
    for (int i = 0; i < 8; ++i) {
      int n = hif * 8 + i;                 // query token within window
      int r = n * VV + vi;                 // faithful torch reshape interleave
      int vout = r >> 4, nout = r & 15;
      size_t row = (size_t)vout * MROWS + (size_t)b_ * 16 + nout;
      ctx[row * CC + head * HD + d] = (_Float16)c0[i];
    }
  }
}

// ---------------- proj GEMM + window reverse + roll + residual -> x2 (f32, d_out) ----
__global__ __launch_bounds__(256) void k_gemm_proj(
    const _Float16* __restrict__ A, const _Float16* __restrict__ Bt,
    const float* __restrict__ bias,
    const float* __restrict__ x0, const float* __restrict__ x1,
    const float* __restrict__ x2, const float* __restrict__ x3,
    float* __restrict__ out) {
  int v = blockIdx.z;
  A  += (size_t)v * MROWS * CC;
  Bt += (size_t)v * CC * CC;
  const float* bi = bias + (size_t)v * CC;
  const float* xp = (v == 0) ? x0 : (v == 1) ? x1 : (v == 2) ? x2 : x3;
  int wid = threadIdx.x >> 5, lane = threadIdx.x & 31;
  int mw = wid >> 2, nw = wid & 3;
  int m0 = blockIdx.y * 64 + mw * 32;
  int n0 = nw * 64;                          // N = 256 -> single block in x
  int lm = lane & 15, hi = lane >> 4, kbase = hi * 8;
  const _Float16* a0p = A + (size_t)(m0 + lm) * CC;
  const _Float16* b0p = Bt + (size_t)(n0 + lm) * CC;
  v8f acc[8];
#pragma unroll
  for (int i = 0; i < 8; ++i) acc[i] = (v8f){0.f,0.f,0.f,0.f,0.f,0.f,0.f,0.f};
  gemm_core_32x64(a0p, a0p + 16 * CC, b0p, b0p + 16 * CC, b0p + 32 * CC, b0p + 48 * CC,
                  CC, kbase, acc);
#pragma unroll
  for (int t = 0; t < 2; ++t) {
#pragma unroll
    for (int j = 0; j < 4; ++j) {
      int col = n0 + j * 16 + lm;
      float bs = bi[col];
      v8f a = acc[t * 4 + j];
#pragma unroll
      for (int i = 0; i < 8; ++i) {
        int row = m0 + t * 16 + hi * 8 + i;  // window-token row
        int b_ = row >> 4, n = row & 15;
        int bb = b_ / (NWH * NWW * ZZ);
        int rem = b_ % (NWH * NWW * ZZ);
        int wh = rem / (NWW * ZZ);
        int rem2 = rem % (NWW * ZZ);
        int ww = rem2 / ZZ, wz = rem2 % ZZ;
        int hh = ((wh << 2) | (n >> 2));
        int wc = ((ww << 2) | (n & 3));
        hh = (hh + 2) & (HH - 1);            // same-sign roll as reference
        wc = (wc + 2) & (WW - 1);
        int l = (hh * WW + wc) * ZZ + wz;
        size_t tok = (size_t)bb * LL + l;
        float val = a[i] + bs + xp[tok * CC + col];   // + shortcut
        out[((size_t)v * BB * LL + tok) * CC + col] = val;
      }
    }
  }
}

// ---------------- FC2 GEMM + residual read-modify-write of d_out ----------------
__global__ __launch_bounds__(256) void k_gemm_fc2(
    const _Float16* __restrict__ A, const _Float16* __restrict__ Bt,
    const float* __restrict__ bias, float* __restrict__ out) {
  int v = blockIdx.z;
  A  += (size_t)v * MROWS * HID_;
  Bt += (size_t)v * CC * HID_;
  const float* bi = bias + (size_t)v * CC;
  int wid = threadIdx.x >> 5, lane = threadIdx.x & 31;
  int mw = wid >> 2, nw = wid & 3;
  int m0 = blockIdx.y * 64 + mw * 32;
  int n0 = nw * 64;                          // N = 256 -> single block in x
  int lm = lane & 15, hi = lane >> 4, kbase = hi * 8;
  const _Float16* a0p = A + (size_t)(m0 + lm) * HID_;
  const _Float16* b0p = Bt + (size_t)(n0 + lm) * HID_;
  v8f acc[8];
#pragma unroll
  for (int i = 0; i < 8; ++i) acc[i] = (v8f){0.f,0.f,0.f,0.f,0.f,0.f,0.f,0.f};
  gemm_core_32x64(a0p, a0p + 16 * HID_, b0p, b0p + 16 * HID_, b0p + 32 * HID_, b0p + 48 * HID_,
                  HID_, kbase, acc);
#pragma unroll
  for (int t = 0; t < 2; ++t) {
#pragma unroll
    for (int j = 0; j < 4; ++j) {
      int col = n0 + j * 16 + lm;
      float bs = bi[col];
      v8f a = acc[t * 4 + j];
#pragma unroll
      for (int i = 0; i < 8; ++i) {
        int row = m0 + t * 16 + hi * 8 + i;
        size_t idx = ((size_t)v * MROWS + row) * CC + col;
        out[idx] = out[idx] + a[i] + bs;     // x2 + mlp
      }
    }
  }
}

// ---------------- host launch ----------------
extern "C" void kernel_launch(void* const* d_in, const int* in_sizes, int n_in,
                              void* d_out, int out_size, void* d_ws, size_t ws_size,
                              hipStream_t stream) {
  (void)in_sizes; (void)n_in; (void)out_size; (void)ws_size;
  const float* x0 = (const float*)d_in[0];
  const float* x1 = (const float*)d_in[1];
  const float* x2 = (const float*)d_in[2];
  const float* x3 = (const float*)d_in[3];
  const float* qkv_w  = (const float*)d_in[4];
  const float* qkv_b  = (const float*)d_in[5];
  const float* proj_w = (const float*)d_in[6];
  const float* proj_b = (const float*)d_in[7];
  const float* n1_g = (const float*)d_in[8];
  const float* n1_b = (const float*)d_in[9];
  const float* n2_g = (const float*)d_in[10];
  const float* n2_b = (const float*)d_in[11];
  const float* fc1_w = (const float*)d_in[12];
  const float* fc1_b = (const float*)d_in[13];
  const float* fc2_w = (const float*)d_in[14];
  const float* fc2_b = (const float*)d_in[15];
  float* out = (float*)d_out;

  char* ws = (char*)d_ws;
  _Float16* qkv_wt  = (_Float16*)(ws);
  _Float16* proj_wt = qkv_wt  + (size_t)VV * 768 * CC;
  _Float16* fc1_wt  = proj_wt + (size_t)VV * CC * CC;
  _Float16* fc2_wt  = fc1_wt  + (size_t)VV * HID_ * CC;
  _Float16* actA = (_Float16*)(ws + (16ull << 20));                 // [V][MROWS][CC]
  _Float16* ctxb = (_Float16*)(ws + (16ull << 20) + (112ull << 20)); // [V][MROWS][CC]
  _Float16* big  = (_Float16*)(ws + (16ull << 20) + (224ull << 20)); // qkv, later h

  dim3 blk(256);
  // 0) weights -> f16 transposed [N,K]
  {
    long n;
    n = (long)VV * CC * 768;  k_wt<<<(unsigned)((n + 255) / 256), blk, 0, stream>>>(qkv_w,  qkv_wt, CC, 768);
    n = (long)VV * CC * CC;   k_wt<<<(unsigned)((n + 255) / 256), blk, 0, stream>>>(proj_w, proj_wt, CC, CC);
    n = (long)VV * CC * HID_; k_wt<<<(unsigned)((n + 255) / 256), blk, 0, stream>>>(fc1_w,  fc1_wt, CC, HID_);
    n = (long)VV * HID_ * CC; k_wt<<<(unsigned)((n + 255) / 256), blk, 0, stream>>>(fc2_w,  fc2_wt, HID_, CC);
  }
  // 1) LN1 + shift + window partition
  k_ln1_window<<<(unsigned)(TOKENS / 8), blk, 0, stream>>>(x0, x1, x2, x3, n1_g, n1_b, actA);
  // 2) QKV GEMM: per v, [MROWS,256] x [256,768]
  k_gemm_f16<<<dim3(768 / 256, MROWS / 64, VV), blk, 0, stream>>>(
      actA, qkv_wt, qkv_b, big, MROWS, 768, CC, 0,
      (size_t)MROWS * CC, (size_t)768 * CC, (size_t)768, (size_t)MROWS * 768);
  // 3) windowed attention (q.kT -> softmax -> .V), one wave per (window,var,head)
  k_attn<<<(unsigned)((long)BWIN * VV * NHD / 8), blk, 0, stream>>>(big, ctxb);
  // 4) proj GEMM + window reverse + roll + residual -> x2 in d_out
  k_gemm_proj<<<dim3(1, MROWS / 64, VV), blk, 0, stream>>>(
      ctxb, proj_wt, proj_b, x0, x1, x2, x3, out);
  // 5) LN2
  k_ln2<<<(unsigned)(TOKENS / 8), blk, 0, stream>>>(out, n2_g, n2_b, actA);
  // 6) FC1 + exact GELU
  k_gemm_f16<<<dim3(HID_ / 256, MROWS / 64, VV), blk, 0, stream>>>(
      actA, fc1_wt, fc1_b, big, MROWS, HID_, CC, 1,
      (size_t)MROWS * CC, (size_t)HID_ * CC, (size_t)HID_, (size_t)MROWS * HID_);
  // 7) FC2 + residual (RMW d_out)
  k_gemm_fc2<<<dim3(1, MROWS / 64, VV), blk, 0, stream>>>(big, fc2_wt, fc2_b, out);
}
